// TemporalGNN_34600256537208
// MI455X (gfx1250) — matrix operand order repaired and checked by reading, not compiled, not run on hardware
//
#include <hip/hip_runtime.h>
#include <cstdint>

typedef __attribute__((ext_vector_type(16))) __bf16 v16bf;
typedef __attribute__((ext_vector_type(8)))  float  v8f;

#define NNODES 50000
#define NEDGES 800000
#define HDIM   128
#define EDGE_BLOCKS 1250   // 8 waves each -> 10000 waves, 5 tiles/wave

__device__ __forceinline__ unsigned short f2bf(float f) {
    unsigned int u = __float_as_uint(f);
    unsigned int r = u + 0x7fffu + ((u >> 16) & 1u);   // round to nearest even
    return (unsigned short)(r >> 16);
}

union ABFrag { uint4 u[2]; v16bf v; };

// ---------------------------------------------------------------- weight swizzle
// Pack fp32 [K x 128] weight into B-fragment order:
// out[((nt*KF + kf)*32 + lane)*8 + i] = bf16pair( B[k][n], B[k+1][n] )
// with half = lane>>4, n = (lane&15) + 16*nt, k = 32*kf + 8*half + 2*(i&3) + 16*(i>>2)
__global__ void k_swizzleB(const float* __restrict__ B, unsigned int* __restrict__ out, int KF) {
    int t = blockIdx.x * blockDim.x + threadIdx.x;
    int total = 8 * KF * 32 * 8;                       // NT=8
    if (t >= total) return;
    int i    =  t        & 7;
    int lane = (t >> 3)  & 31;
    int kf   = (t >> 8)  % KF;
    int nt   =  t / (256 * KF);
    int half = lane >> 4;
    int n    = (lane & 15) + nt * 16;
    int k    = kf * 32 + 8 * half + 2 * (i & 3) + 16 * (i >> 2);
    unsigned short lo = f2bf(B[(size_t)k       * HDIM + n]);
    unsigned short hi = f2bf(B[(size_t)(k + 1) * HDIM + n]);
    out[t] = ((unsigned int)hi << 16) | lo;
}

// ---------------------------------------------------------------- fp32 -> bf16
__global__ void k_cvt_bf16(const float* __restrict__ in, unsigned short* __restrict__ out, int n) {
    int t = blockIdx.x * blockDim.x + threadIdx.x;
    if (t < n) out[t] = f2bf(in[t]);
}

// ---------------------------------------------------------------- degree kernels
__global__ void k_deg_init(float* __restrict__ deg, int n) {
    int t = blockIdx.x * blockDim.x + threadIdx.x;
    if (t < n) deg[t] = 1.0f;                          // self loop
}
__global__ void k_deg_count(const int* __restrict__ dst, float* __restrict__ deg, int e) {
    int t = blockIdx.x * blockDim.x + threadIdx.x;
    if (t < e) atomicAdd(&deg[dst[t]], 1.0f);
}
__global__ void k_deg_rsqrt(float* __restrict__ deg, int n) {
    int t = blockIdx.x * blockDim.x + threadIdx.x;
    if (t < n) deg[t] = rsqrtf(deg[t]);
}

__global__ void k_zero4(float4* __restrict__ p, int n4) {
    int t = blockIdx.x * blockDim.x + threadIdx.x;
    if (t < n4) p[t] = make_float4(0.f, 0.f, 0.f, 0.f);
}

// ---------------------------------------------------------------- node GEMM: C[N,128] = Abf @ Bswz
// One wave per 16-row tile; A (16x128 bf16) held in 4 fragments, reused over 8 N-tiles.
__global__ void __launch_bounds__(128)
k_gemm_node(const unsigned short* __restrict__ Abf,   // [N x 128] bf16 row-major
            const unsigned int*   __restrict__ Bswz,  // swizzled, KF=4, NT=8
            float* __restrict__ C)                    // [N x 128] fp32
{
    int wave = blockIdx.x * (blockDim.x >> 5) + (threadIdx.x >> 5);
    if (wave >= NNODES / 16) return;
    int lane = threadIdx.x & 31;
    int half = lane >> 4, n0 = lane & 15;
    int row0 = wave * 16;

    const uint4* row4 = (const uint4*)(Abf + (size_t)(row0 + n0) * HDIM);
    ABFrag a[4];
#pragma unroll
    for (int j = 0; j < 4; ++j) {
        a[j].u[0] = row4[4 * j + half];
        a[j].u[1] = row4[4 * j + 2 + half];
    }

    v8f acc[8] = {};
#pragma unroll
    for (int nt = 0; nt < 8; ++nt) {
#pragma unroll
        for (int kf = 0; kf < 4; ++kf) {
            ABFrag b;
            const uint4* bp = (const uint4*)(Bswz + ((size_t)(nt * 4 + kf) * 32 + lane) * 8);
            b.u[0] = bp[0];
            b.u[1] = bp[1];
            acc[nt] = __builtin_amdgcn_wmma_f32_16x16x32_bf16(
                false, a[kf].v, false, b.v, (short)0, acc[nt], false, false);
        }
    }
#pragma unroll
    for (int nt = 0; nt < 8; ++nt)
#pragma unroll
        for (int v = 0; v < 8; ++v)
            C[(size_t)(row0 + 8 * half + v) * HDIM + nt * 16 + n0] = acc[nt][v];
}

// ---------------------------------------------------------------- edge scatter: agg[d] += norm * xw[s]
// One wave per edge; 4 floats per lane (128 features).
__global__ void __launch_bounds__(256)
k_scatter(const float* __restrict__ xw, const int* __restrict__ src, const int* __restrict__ dst,
          const float* __restrict__ dinv, float* __restrict__ agg)
{
    int w = (blockIdx.x * blockDim.x + threadIdx.x) >> 5;
    if (w >= NEDGES) return;
    int lane = threadIdx.x & 31;
    int s = src[w], d = dst[w];
    float nrm = dinv[s] * dinv[d];
    float4 v = ((const float4*)(xw + (size_t)s * HDIM))[lane];
    float* ad = agg + (size_t)d * HDIM + lane * 4;
    atomicAdd(ad + 0, v.x * nrm);
    atomicAdd(ad + 1, v.y * nrm);
    atomicAdd(ad + 2, v.z * nrm);
    atomicAdd(ad + 3, v.w * nrm);
}

// ---------------------------------------------------------------- fused self-loop + bias + BN(eval) + ReLU -> bf16
__global__ void k_bn_relu(const float* __restrict__ agg, const float* __restrict__ xw,
                          const float* __restrict__ dinv,
                          const float* __restrict__ b, const float* __restrict__ g,
                          const float* __restrict__ be, const float* __restrict__ m,
                          const float* __restrict__ v, unsigned short* __restrict__ hbf)
{
    int t = blockIdx.x * blockDim.x + threadIdx.x;
    if (t >= NNODES * HDIM) return;
    int i = t >> 7, c = t & 127;
    float di = dinv[i];
    float val = agg[t] + di * di * xw[t] + b[c];
    float y = (val - m[c]) * (g[c] * rsqrtf(v[c] + 1e-5f)) + be[c];
    hbf[t] = f2bf(fmaxf(y, 0.f));
}

// ---------------------------------------------------------------- edge MLP: sigmoid( relu([h_s|h_d]@W1+b1) @ W2 + b2 )
// Persistent waves: epW1 (swizzled, 64KB bf16) staged once per block into LDS;
// each wave grid-strides over 16-edge tiles. K=256, N=128: 8 A-frags, 64 WMMAs/tile,
// B served conflict-free from LDS, next tile's h-rows prefetched to hide gather latency.
__global__ void __launch_bounds__(256)
k_edge_mlp(const unsigned short* __restrict__ hbf, const int* __restrict__ src,
           const int* __restrict__ dst, const unsigned int* __restrict__ Bswz, // KF=8, NT=8
           const float* __restrict__ epb1, const float* __restrict__ epW2,
           const float* __restrict__ epb2, float* __restrict__ out)
{
    __shared__ uint4 lds_b[4096];                      // 64 KB: 64 frags * 32 lanes * 32B
    // ---- stage swizzled weights into LDS (coalesced b128 loads -> ds_store_b128)
    const uint4* bsrc = (const uint4*)Bswz;
#pragma unroll
    for (int it = 0; it < 16; ++it)
        lds_b[threadIdx.x + it * 256] = bsrc[threadIdx.x + it * 256];
    __syncthreads();

    const int lane = threadIdx.x & 31;
    const int half = lane >> 4, n0 = lane & 15;
    const int nWaves = gridDim.x * (blockDim.x >> 5);
    const int gwave  = blockIdx.x * (blockDim.x >> 5) + (threadIdx.x >> 5);
    const float b2 = epb2[0];

    for (int tile = gwave; tile < NEDGES / 16; tile += nWaves) {
        int e = tile * 16 + n0;
        int s = src[e], d = dst[e];

        // prefetch next tile's gather rows (each h-row = 256B)
        int tnext = tile + nWaves;
        if (tnext < NEDGES / 16) {
            int en = tnext * 16 + n0;
            const unsigned short* ps = hbf + (size_t)src[en] * HDIM;
            const unsigned short* pd = hbf + (size_t)dst[en] * HDIM;
            __builtin_prefetch(ps, 0, 1);
            __builtin_prefetch(ps + 64, 0, 1);
            __builtin_prefetch(pd, 0, 1);
            __builtin_prefetch(pd + 64, 0, 1);
        }

        const uint4* rs = (const uint4*)(hbf + (size_t)s * HDIM);
        const uint4* rd = (const uint4*)(hbf + (size_t)d * HDIM);
        ABFrag a[8];
#pragma unroll
        for (int j = 0; j < 4; ++j) {
            a[j].u[0]     = rs[4 * j + half];
            a[j].u[1]     = rs[4 * j + 2 + half];
            a[j + 4].u[0] = rd[4 * j + half];
            a[j + 4].u[1] = rd[4 * j + 2 + half];
        }

        v8f acc[8] = {};
#pragma unroll
        for (int nt = 0; nt < 8; ++nt) {
#pragma unroll
            for (int kf = 0; kf < 8; ++kf) {
                ABFrag b;
                int fi = ((nt * 8 + kf) * 32 + lane) * 2;
                b.u[0] = lds_b[fi];
                b.u[1] = lds_b[fi + 1];
                acc[nt] = __builtin_amdgcn_wmma_f32_16x16x32_bf16(
                    false, a[kf].v, false, b.v, (short)0, acc[nt], false, false);
            }
        }

        // epilogue: z1 = relu(acc + epb1); p = z1 @ epW2 (partial over this lane's 8 cols)
        float p[8] = {0.f, 0.f, 0.f, 0.f, 0.f, 0.f, 0.f, 0.f};
#pragma unroll
        for (int nt = 0; nt < 8; ++nt) {
            int col = nt * 16 + n0;
            float w2 = epW2[col];
            float bb = epb1[col];
#pragma unroll
            for (int v = 0; v < 8; ++v) {
                float t = fmaxf(acc[nt][v] + bb, 0.f);
                p[v] += t * w2;
            }
        }
        // reduce across the 16 lanes of this half-group (xor masks < 16 keep halves disjoint)
#pragma unroll
        for (int off = 1; off < 16; off <<= 1)
#pragma unroll
            for (int v = 0; v < 8; ++v)
                p[v] += __shfl_xor(p[v], off, 32);

        if (n0 == 0) {
#pragma unroll
            for (int v = 0; v < 8; ++v) {
                float z = p[v] + b2;
                out[tile * 16 + 8 * half + v] = 1.f / (1.f + __expf(-z));
            }
        }
    }
}

// ================================================================ host
extern "C" void kernel_launch(void* const* d_in, const int* in_sizes, int n_in,
                              void* d_out, int out_size, void* d_ws, size_t ws_size,
                              hipStream_t stream) {
    const float* x    = (const float*)d_in[0];
    const int*   eidx = (const int*)d_in[1];
    const int*   src  = eidx;
    const int*   dst  = eidx + NEDGES;
    const float* W1   = (const float*)d_in[2];
    const float* b1   = (const float*)d_in[3];
    const float* g1   = (const float*)d_in[4];
    const float* be1  = (const float*)d_in[5];
    const float* m1   = (const float*)d_in[6];
    const float* v1   = (const float*)d_in[7];
    const float* W2   = (const float*)d_in[8];
    const float* b2   = (const float*)d_in[9];
    const float* g2   = (const float*)d_in[10];
    const float* be2  = (const float*)d_in[11];
    const float* m2   = (const float*)d_in[12];
    const float* v2   = (const float*)d_in[13];
    const float* epW1 = (const float*)d_in[14];
    const float* epb1 = (const float*)d_in[15];
    const float* epW2 = (const float*)d_in[16];
    const float* epb2 = (const float*)d_in[17];
    float* out = (float*)d_out;

    // workspace carve-up (256B aligned)
    char* ws = (char*)d_ws;
    size_t off = 0;
    auto carve = [&](size_t bytes) { char* p = ws + off; off = (off + bytes + 255) & ~(size_t)255; return p; };
    float*          deg  = (float*)         carve((size_t)NNODES * 4);
    float*          xw   = (float*)         carve((size_t)NNODES * HDIM * 4);
    float*          agg  = (float*)         carve((size_t)NNODES * HDIM * 4);
    unsigned short* hbf  = (unsigned short*)carve((size_t)NNODES * HDIM * 2);
    unsigned int*   W1s  = (unsigned int*)  carve(8 * 4 * 32 * 8 * 4);
    unsigned int*   W2s  = (unsigned int*)  carve(8 * 4 * 32 * 8 * 4);
    unsigned int*   eps  = (unsigned int*)  carve(8 * 8 * 32 * 8 * 4);
    (void)ws_size; (void)in_sizes; (void)n_in; (void)out_size;

    // weight swizzles (bf16 fragment order)
    k_swizzleB<<<32,  256, 0, stream>>>(W1,   W1s, 4);
    k_swizzleB<<<32,  256, 0, stream>>>(W2,   W2s, 4);
    k_swizzleB<<<64,  256, 0, stream>>>(epW1, eps, 8);

    // x -> bf16 (into hbf, consumed by GEMM1 then overwritten by BN1)
    k_cvt_bf16<<<(NNODES * HDIM + 255) / 256, 256, 0, stream>>>(x, hbf, NNODES * HDIM);

    // degrees: deg = 1 + in-degree; dinv = rsqrt(deg)
    k_deg_init <<<(NNODES + 255) / 256, 256, 0, stream>>>(deg, NNODES);
    k_deg_count<<<(NEDGES + 255) / 256, 256, 0, stream>>>(dst, deg, NEDGES);
    k_deg_rsqrt<<<(NNODES + 255) / 256, 256, 0, stream>>>(deg, NNODES);

    const int gemmBlocks = (NNODES / 16 + 3) / 4;      // 4 waves/block
    const int aggN4 = NNODES * HDIM / 4;

    // ---- layer 1
    k_zero4<<<(aggN4 + 255) / 256, 256, 0, stream>>>((float4*)agg, aggN4);
    k_gemm_node<<<gemmBlocks, 128, 0, stream>>>(hbf, W1s, xw);
    k_scatter<<<NEDGES / 8, 256, 0, stream>>>(xw, src, dst, deg, agg);
    k_bn_relu<<<(NNODES * HDIM + 255) / 256, 256, 0, stream>>>(agg, xw, deg, b1, g1, be1, m1, v1, hbf);

    // ---- layer 2
    k_zero4<<<(aggN4 + 255) / 256, 256, 0, stream>>>((float4*)agg, aggN4);
    k_gemm_node<<<gemmBlocks, 128, 0, stream>>>(hbf, W2s, xw);
    k_scatter<<<NEDGES / 8, 256, 0, stream>>>(xw, src, dst, deg, agg);
    k_bn_relu<<<(NNODES * HDIM + 255) / 256, 256, 0, stream>>>(agg, xw, deg, b2, g2, be2, m2, v2, hbf);

    // ---- edge predictor: persistent waves, LDS-staged weights
    k_edge_mlp<<<EDGE_BLOCKS, 256, 0, stream>>>(hbf, src, dst, eps, epb1, epW2, epb2, out);
}
